// CrossScaleAttention_6459630813246
// MI455X (gfx1250) — compile-verified, matrix operand order checked by reading, not run on hardware
//
#include <hip/hip_runtime.h>
#include <hip/hip_bf16.h>

// ---------------------------------------------------------------------------
// CrossScaleAttention for MI455X (gfx1250, wave32, WMMA).
// Heavy math (correlation GEMM + aggregation GEMM) runs on
// v_wmma_f32_16x16x32_f16 with fp32 accumulators; LDS tiles are staged with
// GLOBAL_LOAD_ASYNC_TO_LDS_B128 (ASYNCcnt) when the builtin exists, with a
// batched sync fallback. Double-buffered LDS overlaps fetch with WMMA.
// ---------------------------------------------------------------------------

typedef __attribute__((ext_vector_type(16))) _Float16 v16h;
typedef __attribute__((ext_vector_type(8)))  float    v8f;
typedef int v4i_vs __attribute__((vector_size(16)));   // b128 payload type

#define BDIM 256

#if defined(__HIP_DEVICE_COMPILE__) && __has_builtin(__builtin_amdgcn_global_load_async_to_lds_b128)
#define USE_ASYNC_LDS 1
#else
#define USE_ASYNC_LDS 0
#endif

__device__ __forceinline__ void gl_async_b128(const _Float16* g, _Float16* l) {
#if USE_ASYNC_LDS
    __builtin_amdgcn_global_load_async_to_lds_b128(
        (__attribute__((address_space(1))) v4i_vs*)(void*)g,
        (__attribute__((address_space(3))) v4i_vs*)(void*)l, 0, 0);
#else
    (void)g; (void)l;
#endif
}

__device__ __forceinline__ void wait_async0() {
#if USE_ASYNC_LDS
#if __has_builtin(__builtin_amdgcn_s_wait_asynccnt)
    __builtin_amdgcn_s_wait_asynccnt(0);
#else
    asm volatile("s_wait_asynccnt 0x0" ::: "memory");
#endif
#endif
}

// ------------------------- small elementwise kernels ------------------------

// out[o,p] = prelu(sum_c w[o,c]*in[c,p] + b[o]); grid = (ceil(P/256), O)
__global__ __launch_bounds__(BDIM)
void conv1x1_prelu(const float* __restrict__ in, const float* __restrict__ w,
                   const float* __restrict__ b, const float* __restrict__ a,
                   float* __restrict__ out, int Cin, int P) {
    int p = blockIdx.x * BDIM + threadIdx.x;
    int o = blockIdx.y;
    if (p >= P) return;
    float s = b[o];
    #pragma unroll 8
    for (int c = 0; c < Cin; ++c)
        s += w[o * Cin + c] * in[(size_t)c * P + p];
    float alpha = a[0];
    out[(size_t)o * P + p] = (s >= 0.f) ? s : alpha * s;
}

// Query patches: Ap[p=(Y*96+X), k=c*9+dy*3+dx] = mi[c, Y+dy-1, X+dx-1] (f16)
__global__ __launch_bounds__(BDIM)
void build_query_patches(const float* __restrict__ mi, _Float16* __restrict__ Ap) {
    int idx = blockIdx.x * BDIM + threadIdx.x;           // 9216*288
    if (idx >= 9216 * 288) return;
    int p = idx / 288, k = idx % 288;
    int Yq = p / 96, Xq = p % 96;
    int c = k / 9, r = k % 9, dy = r / 3, dx = r % 3;
    int yy = Yq + dy - 1, xx = Xq + dx - 1;
    float v = (yy >= 0 && yy < 96 && xx >= 0 && xx < 96)
                  ? mi[(c * 96 + yy) * 96 + xx] : 0.f;
    Ap[idx] = (_Float16)v;
}

// Key patches, L2-normalized per patch: Wn[l, k] (f16), l over 48x48, k=c*9+dy*3+dx
__global__ __launch_bounds__(BDIM)
void build_key_patches(const float* __restrict__ rf, _Float16* __restrict__ Wn) {
    int l = blockIdx.x * BDIM + threadIdx.x;
    if (l >= 2304) return;
    int yl = l / 48, xl = l % 48;
    float ss = 0.f;
    for (int c = 0; c < 32; ++c)
        for (int dy = 0; dy < 3; ++dy)
            for (int dx = 0; dx < 3; ++dx) {
                int yy = yl + dy - 1, xx = xl + dx - 1;
                float v = (yy >= 0 && yy < 48 && xx >= 0 && xx < 48)
                              ? rf[(c * 48 + yy) * 48 + xx] : 0.f;
                ss += v * v;
            }
    float inv = 1.f / fmaxf(sqrtf(ss), 1e-4f);
    for (int c = 0; c < 32; ++c)
        for (int dy = 0; dy < 3; ++dy)
            for (int dx = 0; dx < 3; ++dx) {
                int yy = yl + dy - 1, xx = xl + dx - 1;
                float v = (yy >= 0 && yy < 48 && xx >= 0 && xx < 48)
                              ? rf[(c * 48 + yy) * 48 + xx] : 0.f;
                Wn[(size_t)l * 288 + c * 9 + dy * 3 + dx] = (_Float16)(v * inv);
            }
}

// Value patches, transposed layout for NT GEMM: Rwt[n=c*9+dy*3+dx, l] (f16),
// padded to n in [0,640) with zeros.
__global__ __launch_bounds__(BDIM)
void build_value_patches(const float* __restrict__ ew, _Float16* __restrict__ Rwt) {
    int idx = blockIdx.x * BDIM + threadIdx.x;           // 640*2304
    if (idx >= 640 * 2304) return;
    int n = idx / 2304, l = idx % 2304;
    float v = 0.f;
    if (n < 576) {
        int c = n / 9, r = n % 9, dy = r / 3, dx = r % 3;
        int yl = l / 48, xl = l % 48;
        int yy = yl + dy - 1, xx = xl + dx - 1;
        if (yy >= 0 && yy < 48 && xx >= 0 && xx < 48)
            v = ew[(c * 48 + yy) * 48 + xx];
    }
    Rwt[idx] = (_Float16)v;
}

// ------------------------------- WMMA GEMM ---------------------------------
// C[M,N] = A[M,K] * Bt[N,K]^T ; f16 in, fp32 out. M%128==0, N%128==0, K%32==0.

#define TBM 128
#define TBN 128
#define TBK 32

union FragU { v16h v; uint4 q[2]; };

// Fragment per documented gfx1250 16-bit layout: lane%16 = row (M or N),
// lane/16 selects K-subhalf; elems 0..7 = K[ko..ko+7], 8..15 = K[16+ko..23+ko].
__device__ __forceinline__ v16h load_frag(const _Float16* base, int ld, int lane) {
    int r  = lane & 15;
    int ko = (lane >> 4) * 8;
    FragU f;
    f.q[0] = *(const uint4*)(base + r * ld + ko);
    f.q[1] = *(const uint4*)(base + r * ld + 16 + ko);
    return f.v;
}

__global__ __launch_bounds__(BDIM)
void wmma_gemm_nt(const _Float16* __restrict__ A, const _Float16* __restrict__ Bt,
                  float* __restrict__ C, int M, int N, int K) {
    __shared__ __align__(16) _Float16 As[2][TBM * TBK];
    __shared__ __align__(16) _Float16 Bs[2][TBN * TBK];

    const int tid  = threadIdx.x;
    const int lane = tid & 31;
    const int wave = tid >> 5;       // 0..7
    const int wrow = wave & 3;       // 4 wave-rows  -> 32 rows each
    const int wcol = wave >> 2;      // 2 wave-cols  -> 64 cols each
    const int m0 = blockIdx.y * TBM;
    const int n0 = blockIdx.x * TBN;

    v8f zero = {};
    v8f acc[2][4];
    #pragma unroll
    for (int i = 0; i < 2; ++i)
        #pragma unroll
        for (int j = 0; j < 4; ++j) acc[i][j] = zero;

#if !USE_ASYNC_LDS
    uint4 ra[2], rb[2];
#endif

    const int nk = K / TBK;

    // ---- stage K-tile 0 into buffer 0 ----
    #pragma unroll
    for (int i = 0; i < 2; ++i) {
        int idx = tid + i * BDIM;
        int row = idx >> 2, col = (idx & 3) * 8;
        const _Float16* ga = A  + (size_t)(m0 + row) * K + col;
        const _Float16* gb = Bt + (size_t)(n0 + row) * K + col;
#if USE_ASYNC_LDS
        gl_async_b128(ga, &As[0][row * TBK + col]);
        gl_async_b128(gb, &Bs[0][row * TBK + col]);
#else
        ra[i] = *(const uint4*)ga;
        rb[i] = *(const uint4*)gb;
#endif
    }
#if USE_ASYNC_LDS
    wait_async0();
#else
    #pragma unroll
    for (int i = 0; i < 2; ++i) {
        int idx = tid + i * BDIM;
        int row = idx >> 2, col = (idx & 3) * 8;
        *(uint4*)&As[0][row * TBK + col] = ra[i];
        *(uint4*)&Bs[0][row * TBK + col] = rb[i];
    }
#endif
    __syncthreads();

    for (int kt = 0; kt < nk; ++kt) {
        const int cur = kt & 1;
        const int nxt = cur ^ 1;
        const bool have_next = (kt + 1) < nk;

        // ---- issue fetch of K-tile kt+1 into the other buffer ----
        if (have_next) {
            const int k0 = (kt + 1) * TBK;
            #pragma unroll
            for (int i = 0; i < 2; ++i) {
                int idx = tid + i * BDIM;
                int row = idx >> 2, col = (idx & 3) * 8;
                const _Float16* ga = A  + (size_t)(m0 + row) * K + k0 + col;
                const _Float16* gb = Bt + (size_t)(n0 + row) * K + k0 + col;
#if USE_ASYNC_LDS
                gl_async_b128(ga, &As[nxt][row * TBK + col]);
                gl_async_b128(gb, &Bs[nxt][row * TBK + col]);
#else
                ra[i] = *(const uint4*)ga;
                rb[i] = *(const uint4*)gb;
#endif
            }
        }

        // ---- WMMA on current buffer ----
        v16h af[2], bf[4];
        #pragma unroll
        for (int i = 0; i < 2; ++i)
            af[i] = load_frag(&As[cur][(wrow * 32 + i * 16) * TBK], TBK, lane);
        #pragma unroll
        for (int j = 0; j < 4; ++j)
            bf[j] = load_frag(&Bs[cur][(wcol * 64 + j * 16) * TBK], TBK, lane);

        #pragma unroll
        for (int i = 0; i < 2; ++i)
            #pragma unroll
            for (int j = 0; j < 4; ++j)
                acc[i][j] = __builtin_amdgcn_wmma_f32_16x16x32_f16(
                    false, af[i], false, bf[j], (short)0, acc[i][j], false, false);

        // ---- complete the fetch of tile kt+1, then one barrier ----
        if (have_next) {
#if USE_ASYNC_LDS
            wait_async0();
#else
            #pragma unroll
            for (int i = 0; i < 2; ++i) {
                int idx = tid + i * BDIM;
                int row = idx >> 2, col = (idx & 3) * 8;
                *(uint4*)&As[nxt][row * TBK + col] = ra[i];
                *(uint4*)&Bs[nxt][row * TBK + col] = rb[i];
            }
#endif
        }
        __syncthreads();
    }

    // D layout: elem e of v8f -> row = e + 8*(lane/16), col = lane%16
    const int cl = lane & 15;
    const int ch = lane >> 4;
    #pragma unroll
    for (int i = 0; i < 2; ++i)
        #pragma unroll
        for (int j = 0; j < 4; ++j) {
            int rbase = m0 + wrow * 32 + i * 16 + ch * 8;
            int cbase = n0 + wcol * 64 + j * 16 + cl;
            #pragma unroll
            for (int e = 0; e < 8; ++e)
                C[(size_t)(rbase + e) * N + cbase] = acc[i][j][e];
        }
}

// ------------------------------- softmax -----------------------------------
// One block per row of Y[9216, 2304]; writes f16 probabilities to Ys.
__global__ __launch_bounds__(BDIM)
void softmax_rows(const float* __restrict__ Y, _Float16* __restrict__ Ys, int Nc) {
    __shared__ float red[BDIM];
    const int tid = threadIdx.x;
    const float* row = Y + (size_t)blockIdx.x * Nc;

    float m = -3.4e38f;
    for (int i = tid; i < Nc; i += BDIM) m = fmaxf(m, row[i]);
    red[tid] = m; __syncthreads();
    for (int s = BDIM / 2; s > 0; s >>= 1) {
        if (tid < s) red[tid] = fmaxf(red[tid], red[tid + s]);
        __syncthreads();
    }
    m = red[0]; __syncthreads();

    float e[9];                       // 2304 / 256 == 9 exactly
    float sum = 0.f;
    int c = 0;
    for (int i = tid; i < Nc; i += BDIM) {
        float v = __expf(10.f * (row[i] - m));
        e[c++] = v; sum += v;
    }
    red[tid] = sum; __syncthreads();
    for (int s = BDIM / 2; s > 0; s >>= 1) {
        if (tid < s) red[tid] += red[tid + s];
        __syncthreads();
    }
    float inv = 1.f / red[0];
    c = 0;
    for (int i = tid; i < Nc; i += BDIM)
        Ys[(size_t)blockIdx.x * Nc + i] = (_Float16)(e[c++] * inv);
}

// ---------------------- transpose-conv as a gather -------------------------
// out[c, Yo, Xo] = (1/6) * sum over valid (dy,dx) of P[(y*96+x), c*9+dy*3+dx]
// with y=(Yo+1-dy)/2, x=(Xo+1-dx)/2 integral & in range. P stride = 640.
__global__ __launch_bounds__(BDIM)
void gather_output(const float* __restrict__ P, float* __restrict__ out) {
    const int HW = 191 * 191;
    int idx = blockIdx.x * BDIM + threadIdx.x;
    if (idx >= 64 * HW) return;
    int c = idx / HW, rem = idx % HW;
    int Yo = rem / 191, Xo = rem % 191;

    int dys[2], dxs[2], ny, nx;
    if (Yo & 1) { dys[0] = 0; dys[1] = 2; ny = 2; } else { dys[0] = 1; ny = 1; }
    if (Xo & 1) { dxs[0] = 0; dxs[1] = 2; nx = 2; } else { dxs[0] = 1; nx = 1; }

    float s = 0.f;
    for (int i = 0; i < ny; ++i) {
        int dy = dys[i], y = (Yo + 1 - dy) >> 1;
        for (int j = 0; j < nx; ++j) {
            int dx = dxs[j], x = (Xo + 1 - dx) >> 1;
            s += P[(size_t)(y * 96 + x) * 640 + c * 9 + dy * 3 + dx];
        }
    }
    out[idx] = s * (1.f / 6.f);
}

// ------------------------------- launcher ----------------------------------

extern "C" void kernel_launch(void* const* d_in, const int* in_sizes, int n_in,
                              void* d_out, int out_size, void* d_ws, size_t ws_size,
                              hipStream_t stream) {
    const float* input = (const float*)d_in[0];   // [4,64,96,96]
    const float* small = (const float*)d_in[1];   // [4,64,48,48]
    const float* w_m1  = (const float*)d_in[2];
    const float* b_m1  = (const float*)d_in[3];
    const float* a_m1  = (const float*)d_in[4];
    const float* w_m2  = (const float*)d_in[5];
    const float* b_m2  = (const float*)d_in[6];
    const float* a_m2  = (const float*)d_in[7];
    const float* w_as  = (const float*)d_in[8];
    const float* b_as  = (const float*)d_in[9];
    const float* a_as  = (const float*)d_in[10];
    float* out = (float*)d_out;

    char* wp = (char*)d_ws;
    auto alloc = [&](size_t bytes) {
        void* r = (void*)wp;
        wp += (bytes + 255) & ~(size_t)255;
        return r;
    };
    float*     mi  = (float*)    alloc((size_t)32 * 9216 * 4);  // match_input
    float*     ew  = (float*)    alloc((size_t)64 * 2304 * 4);  // embed_w
    float*     rf  = (float*)    alloc((size_t)32 * 2304 * 4);  // ref
    _Float16*  Ap  = (_Float16*) alloc((size_t)9216 * 288 * 2); // query patches
    _Float16*  Wn  = (_Float16*) alloc((size_t)2304 * 288 * 2); // norm key patches
    _Float16*  Rwt = (_Float16*) alloc((size_t)640 * 2304 * 2); // value patches (N-padded)
    _Float16*  Ys  = (_Float16*) alloc((size_t)9216 * 2304 * 2);// softmax probs
    float*     Y   = (float*)    alloc((size_t)9216 * 2304 * 4);// logits
    float*     Pm  = Y;  // P[9216,640] aliases Y (Y dead after softmax_rows)

    for (int b = 0; b < 4; ++b) {
        const float* inp = input + (size_t)b * 64 * 9216;
        const float* sml = small + (size_t)b * 64 * 2304;
        float* outb = out + (size_t)b * 64 * 191 * 191;

        conv1x1_prelu<<<dim3(36, 32), BDIM, 0, stream>>>(inp, w_m1, b_m1, a_m1, mi, 64, 9216);
        conv1x1_prelu<<<dim3(9, 64),  BDIM, 0, stream>>>(sml, w_as, b_as, a_as, ew, 64, 2304);
        conv1x1_prelu<<<dim3(9, 32),  BDIM, 0, stream>>>(sml, w_m2, b_m2, a_m2, rf, 64, 2304);

        build_query_patches<<<(9216 * 288 + BDIM - 1) / BDIM, BDIM, 0, stream>>>(mi, Ap);
        build_key_patches  <<<(2304 + BDIM - 1) / BDIM,       BDIM, 0, stream>>>(rf, Wn);
        build_value_patches<<<(640 * 2304 + BDIM - 1) / BDIM, BDIM, 0, stream>>>(ew, Rwt);

        // GEMM1: Y[9216,2304] = Ap[9216,288] x Wn[2304,288]^T
        wmma_gemm_nt<<<dim3(2304 / TBN, 9216 / TBM), BDIM, 0, stream>>>(
            Ap, Wn, Y, 9216, 2304, 288);

        softmax_rows<<<9216, BDIM, 0, stream>>>(Y, Ys, 2304);

        // GEMM2: P[9216,640] = Ys[9216,2304] x Rwt[640,2304]^T
        wmma_gemm_nt<<<dim3(640 / TBN, 9216 / TBM), BDIM, 0, stream>>>(
            Ys, Rwt, Pm, 9216, 640, 2304);

        gather_output<<<(64 * 191 * 191 + BDIM - 1) / BDIM, BDIM, 0, stream>>>(Pm, outb);
    }
}
